// RGN2_Naive_55070070670297
// MI455X (gfx1250) — compile-verified, matrix operand order checked by reading, not run on hardware
//
#include <hip/hip_runtime.h>

// ---------------------------------------------------------------------------
// Problem constants (match reference)
// ---------------------------------------------------------------------------
#define BATCH   8
#define SEQ     2048
#define EMB     1280
#define HDIM    256
#define GDIM    1024            // 4*H
#define ROWS    (BATCH * SEQ)   // 16384
#define NWG     16              // persistent workgroups in the scan

typedef __bf16 bf16;
typedef __attribute__((ext_vector_type(16))) __bf16 v16bf;
typedef __attribute__((ext_vector_type(8)))  __bf16 v8bf;
typedef __attribute__((ext_vector_type(8)))  float  v8f;

// ---------------------------------------------------------------------------
// Helpers
// ---------------------------------------------------------------------------
static __device__ __forceinline__ unsigned short f32_to_bf16_bits(float f) {
    unsigned u = __float_as_uint(f);
    u += 0x7FFFu + ((u >> 16) & 1u);   // round-to-nearest-even
    return (unsigned short)(u >> 16);
}
static __device__ __forceinline__ float bf16_bits_to_f32(unsigned short s) {
    return __uint_as_float(((unsigned)s) << 16);
}
static __device__ __forceinline__ float sigmoidf_(float x) {
    return 1.0f / (1.0f + __expf(-x));
}

// Build a 16-element bf16 WMMA fragment for this lane.
// ISA 16-bit A/B layout: lanes 0-15 hold K = {kb..kb+7, kb+16..kb+23} with
// kb=0; lanes 16-31 the same with kb=8.  rowPtr points at (row, k0+kb).
static __device__ __forceinline__ v16bf load_frag_bf16(const unsigned short* rowPtr) {
    v8bf lo = *(const v8bf*)(rowPtr);
    v8bf hi = *(const v8bf*)(rowPtr + 16);
    v16bf v;
#pragma unroll
    for (int i = 0; i < 8; ++i) { v[i] = lo[i]; v[i + 8] = hi[i]; }
    return v;
}

// Device-scope sense-reversing grid barrier.  bar[0]=count, bar[1]=generation.
// Release fence before / acquire fence after are executed by ALL threads so
// every lane's surrounding global accesses are ordered (global_wb/global_inv).
static __device__ __forceinline__ void grid_barrier(unsigned* bar, unsigned nwg) {
    __builtin_amdgcn_fence(__ATOMIC_RELEASE, "agent");
    __syncthreads();
    if (threadIdx.x == 0) {
        unsigned gen = __hip_atomic_load(&bar[1], __ATOMIC_RELAXED, __HIP_MEMORY_SCOPE_AGENT);
        unsigned arrived = __hip_atomic_fetch_add(&bar[0], 1u, __ATOMIC_ACQ_REL, __HIP_MEMORY_SCOPE_AGENT);
        if (arrived == nwg - 1) {
            __hip_atomic_store(&bar[0], 0u, __ATOMIC_RELAXED, __HIP_MEMORY_SCOPE_AGENT);
            __hip_atomic_store(&bar[1], gen + 1u, __ATOMIC_RELEASE, __HIP_MEMORY_SCOPE_AGENT);
        } else {
            while (__hip_atomic_load(&bar[1], __ATOMIC_ACQUIRE, __HIP_MEMORY_SCOPE_AGENT) == gen) {
                __builtin_amdgcn_s_sleep(1);
            }
        }
    }
    __syncthreads();
    __builtin_amdgcn_fence(__ATOMIC_ACQUIRE, "agent");
}

// ---------------------------------------------------------------------------
// Setup kernels
// ---------------------------------------------------------------------------
__global__ void k_cvt_bf16(const float* __restrict__ src,
                           unsigned short* __restrict__ dst, int n) {
    int i = blockIdx.x * blockDim.x + threadIdx.x;
    int stride = gridDim.x * blockDim.x;
    for (; i < n; i += stride) dst[i] = f32_to_bf16_bits(src[i]);
}

__global__ void k_bias_sum(const float* __restrict__ a, const float* __restrict__ b,
                           float* __restrict__ out, int n) {
    int i = blockIdx.x * blockDim.x + threadIdx.x;
    if (i < n) out[i] = a[i] + b[i];
}

__global__ void k_init_barrier(unsigned* bar) {
    if (threadIdx.x == 0) { bar[0] = 0u; bar[1] = 0u; }
}

// ---------------------------------------------------------------------------
// Input-projection GEMM:  Gin[row, col] = A[row,:] . W[col,:] + bsum[col]
//   A: ROWS x K (bf16), W: GDIM x K (bf16).  2 M-tiles x 4 N-tiles per wave
//   (8 accumulators) to cut L2 traffic; K-loop of 16x16x32 bf16 WMMA.
// ---------------------------------------------------------------------------
#define GEMM_MT 2
#define GEMM_NT 4
__global__ void __launch_bounds__(256)
k_gemm_gin(const unsigned short* __restrict__ A,
           const unsigned short* __restrict__ W,
           const float* __restrict__ bsum,
           float* __restrict__ Gin, int K) {
    const unsigned wave = threadIdx.x >> 5;
    const unsigned lane = threadIdx.x & 31u;
    // job grid: (ROWS/16/GEMM_MT) x (GDIM/16/GEMM_NT) = 512 x 16 wave jobs
    const unsigned job = blockIdx.x * (blockDim.x >> 5) + wave;
    const unsigned tNg = job & 15u;        // 16 N-groups (each 4 tiles)
    const unsigned tMg = job >> 4;         // 512 M-groups (each 2 tiles)

    const unsigned r  = lane & 15u;        // row-within-tile
    const unsigned kb = (lane >> 4) << 3;  // 0 or 8 (lane-half K offset)

    const unsigned short* ap[GEMM_MT];
    const unsigned short* bp[GEMM_NT];
#pragma unroll
    for (int mt = 0; mt < GEMM_MT; ++mt)
        ap[mt] = A + (size_t)(tMg * (16u * GEMM_MT) + mt * 16u + r) * (size_t)K + kb;
#pragma unroll
    for (int nt = 0; nt < GEMM_NT; ++nt)
        bp[nt] = W + (size_t)(tNg * (16u * GEMM_NT) + nt * 16u + r) * (size_t)K + kb;

    v8f acc[GEMM_MT][GEMM_NT];
#pragma unroll
    for (int mt = 0; mt < GEMM_MT; ++mt)
#pragma unroll
        for (int nt = 0; nt < GEMM_NT; ++nt) acc[mt][nt] = (v8f){};

    for (int k0 = 0; k0 < K; k0 += 32) {
        v16bf av[GEMM_MT], bv[GEMM_NT];
#pragma unroll
        for (int mt = 0; mt < GEMM_MT; ++mt) av[mt] = load_frag_bf16(ap[mt] + k0);
#pragma unroll
        for (int nt = 0; nt < GEMM_NT; ++nt) bv[nt] = load_frag_bf16(bp[nt] + k0);
#pragma unroll
        for (int mt = 0; mt < GEMM_MT; ++mt)
#pragma unroll
            for (int nt = 0; nt < GEMM_NT; ++nt)
                acc[mt][nt] = __builtin_amdgcn_wmma_f32_16x16x32_bf16(
                    false, av[mt], false, bv[nt], (short)0, acc[mt][nt], false, false);
    }

    // D layout: VGPR v holds M = v + 8*(lane>=16), N = lane&15
    const unsigned moff = (lane >> 4) << 3;
#pragma unroll
    for (int nt = 0; nt < GEMM_NT; ++nt) {
        const unsigned col = tNg * (16u * GEMM_NT) + nt * 16u + (lane & 15u);
        const float bs = bsum[col];
#pragma unroll
        for (int mt = 0; mt < GEMM_MT; ++mt) {
#pragma unroll
            for (int v = 0; v < 8; ++v) {
                unsigned row = tMg * (16u * GEMM_MT) + mt * 16u + moff + (unsigned)v;
                Gin[(size_t)row * GDIM + col] = acc[mt][nt][v] + bs;
            }
        }
    }
}

// ---------------------------------------------------------------------------
// Persistent masked-LSTM scan.  16 workgroups x 128 threads (4 waves).
// Block b owns hidden units [16*b, 16*b+16); wave g in {0..3} computes gate g.
// Ping-pong h buffers (read buf[t&1], write buf[(t+1)&1]) so one grid barrier
// per step is race-free.  Gin of step t+1 is prefetched during step t.
// ---------------------------------------------------------------------------
__global__ void __launch_bounds__(128)
k_scan(const float* __restrict__ Gin,              // ROWS x GDIM f32
       const unsigned short* __restrict__ Whh,     // GDIM x HDIM bf16
       const unsigned char* __restrict__ mask,     // BATCH x SEQ
       unsigned short* __restrict__ hbuf,          // 2 x 16 x HDIM bf16 (rows 8..15 zero)
       unsigned short* __restrict__ Hout,          // ROWS x HDIM bf16
       unsigned* __restrict__ bar) {
    __shared__ float gsm[4][8][16];   // [gate][batch][unit]
    __shared__ float csm[8][16];
    __shared__ float hsm[8][16];

    const unsigned g    = threadIdx.x >> 5;   // wave index == gate index
    const unsigned lane = threadIdx.x & 31u;
    const unsigned j0   = blockIdx.x * 16u;   // hidden-unit slice base

    // h0 = c0 = 0; zero both ping-pong buffers (incl. padding rows 8..15).
    if (threadIdx.x < 16u) {
#pragma unroll
        for (int mr = 0; mr < 32; ++mr) hbuf[mr * HDIM + j0 + threadIdx.x] = 0;
    }
    {
        unsigned b = threadIdx.x >> 4, jj = threadIdx.x & 15u;
        if (b < 8u) { csm[b][jj] = 0.0f; hsm[b][jj] = 0.0f; }
    }
    grid_barrier(bar, NWG);

    const unsigned r  = lane & 15u;
    const unsigned kb = (lane >> 4) << 3;
    const unsigned gaterow = g * HDIM + j0 + r;       // row of W_hh / gate column
    const unsigned short* bp  = Whh + (size_t)gaterow * HDIM + kb;
    const unsigned short* ap0 = hbuf + (size_t)r * HDIM + kb;            // buffer 0
    const unsigned short* ap1 = ap0 + (size_t)16 * HDIM;                 // buffer 1

    for (int t = 0; t < SEQ; ++t) {
        const unsigned short* ap = (t & 1) ? ap1 : ap0;
        unsigned short* hw = hbuf + (size_t)((t + 1) & 1) * 16u * HDIM;

        // gates_slice = h_prev(16x256, rows 8..15 zero) @ Whh_slice^T
        v8f acc = {};
#pragma unroll
        for (int k0 = 0; k0 < HDIM; k0 += 32) {
            v16bf av = load_frag_bf16(ap + k0);
            v16bf bv = load_frag_bf16(bp + k0);
            acc = __builtin_amdgcn_wmma_f32_16x16x32_bf16(
                false, av, false, bv, (short)0, acc, false, false);
        }

        // Prefetch next step's Gin cachelines while WMMAs drain.
        if ((t + 1) < SEQ && lane < 8u) {
            __builtin_prefetch(
                &Gin[((size_t)lane * SEQ + (size_t)(t + 1)) * GDIM + g * HDIM + j0], 0, 3);
        }

        // lanes 0..15 hold batch rows 0..7 in acc[0..7]; add Gin, stash in LDS
        if (lane < 16u) {
            unsigned col = g * HDIM + j0 + lane;
#pragma unroll
            for (int v = 0; v < 8; ++v) {
                float gv = acc[v] + Gin[((size_t)v * SEQ + (size_t)t) * GDIM + col];
                gsm[g][v][lane] = gv;
            }
        }
        __syncthreads();

        // 128 threads <-> 8 batches x 16 units: LSTM cell update
        {
            unsigned b = threadIdx.x >> 4, jj = threadIdx.x & 15u;
            float iv = sigmoidf_(gsm[0][b][jj]);
            float fv = sigmoidf_(gsm[1][b][jj]);
            float gv = tanhf(gsm[2][b][jj]);
            float ov = sigmoidf_(gsm[3][b][jj]);
            float c_new = fv * csm[b][jj] + iv * gv;
            float h_new = ov * tanhf(c_new);
            if (mask[(size_t)b * SEQ + (size_t)t]) {
                csm[b][jj] = c_new;
                hsm[b][jj] = h_new;
            }
            unsigned short hb = f32_to_bf16_bits(hsm[b][jj]);
            hw[b * HDIM + j0 + jj] = hb;                                    // next h
            Hout[((size_t)b * SEQ + (size_t)t) * HDIM + j0 + jj] = hb;      // record
        }
        grid_barrier(bar, NWG);
    }
}

// ---------------------------------------------------------------------------
// MLP head: out = silu(h @ W1^T + b1) @ W2^T + b2   (H=256 -> 32 -> 4)
// One row per thread; W1/W2 are uniform across the wave (scalar-load friendly).
// ---------------------------------------------------------------------------
__global__ void k_mlp(const unsigned short* __restrict__ H,
                      const float* __restrict__ W1, const float* __restrict__ b1,
                      const float* __restrict__ W2, const float* __restrict__ b2,
                      float* __restrict__ out) {
    int r = blockIdx.x * blockDim.x + threadIdx.x;
    if (r >= ROWS) return;
    const unsigned short* h = H + (size_t)r * HDIM;

    float acc[32];
#pragma unroll
    for (int j = 0; j < 32; ++j) acc[j] = b1[j];
    for (int k = 0; k < HDIM; ++k) {
        float hk = bf16_bits_to_f32(h[k]);
#pragma unroll
        for (int j = 0; j < 32; ++j) acc[j] = fmaf(hk, W1[j * HDIM + k], acc[j]);
    }
#pragma unroll
    for (int j = 0; j < 32; ++j) acc[j] = acc[j] * sigmoidf_(acc[j]);   // silu
#pragma unroll
    for (int o = 0; o < 4; ++o) {
        float s = b2[o];
#pragma unroll
        for (int j = 0; j < 32; ++j) s = fmaf(acc[j], W2[o * 32 + j], s);
        out[(size_t)r * 4 + o] = s;
    }
}

// ---------------------------------------------------------------------------
// Host orchestration
// ---------------------------------------------------------------------------
extern "C" void kernel_launch(void* const* d_in, const int* in_sizes, int n_in,
                              void* d_out, int out_size, void* d_ws, size_t ws_size,
                              hipStream_t stream) {
    (void)in_sizes; (void)n_in; (void)out_size; (void)ws_size;

    const float* x = (const float*)d_in[0];
    const unsigned char* mask = (const unsigned char*)d_in[1];
    const float* Wih[3] = {(const float*)d_in[2], (const float*)d_in[6],  (const float*)d_in[10]};
    const float* Whh[3] = {(const float*)d_in[3], (const float*)d_in[7],  (const float*)d_in[11]};
    const float* bih[3] = {(const float*)d_in[4], (const float*)d_in[8],  (const float*)d_in[12]};
    const float* bhh[3] = {(const float*)d_in[5], (const float*)d_in[9],  (const float*)d_in[13]};
    const float* W1 = (const float*)d_in[14];
    const float* b1 = (const float*)d_in[15];
    const float* W2 = (const float*)d_in[16];
    const float* b2 = (const float*)d_in[17];

    const int inK[3] = {EMB, HDIM, HDIM};

    // ---- carve workspace (256B aligned chunks) ----
    char* p = (char*)d_ws;
    auto carve = [&](size_t bytes) -> char* {
        char* r = p;
        p += (bytes + 255) & ~(size_t)255;
        return r;
    };
    unsigned short* xbf = (unsigned short*)carve((size_t)ROWS * EMB * 2);
    unsigned short* wihbf[3];
    unsigned short* whhbf[3];
    float* bsum[3];
    for (int k = 0; k < 3; ++k) {
        wihbf[k] = (unsigned short*)carve((size_t)GDIM * inK[k] * 2);
        whhbf[k] = (unsigned short*)carve((size_t)GDIM * HDIM * 2);
        bsum[k]  = (float*)carve((size_t)GDIM * 4);
    }
    float* Gin = (float*)carve((size_t)ROWS * GDIM * 4);
    unsigned short* HoutA = (unsigned short*)carve((size_t)ROWS * HDIM * 2);
    unsigned short* HoutB = (unsigned short*)carve((size_t)ROWS * HDIM * 2);
    unsigned short* hbuf  = (unsigned short*)carve((size_t)2 * 16 * HDIM * 2);
    unsigned* bar = (unsigned*)carve(256);

    // ---- setup: conversions, bias sums, barrier init ----
    k_cvt_bf16<<<2048, 256, 0, stream>>>(x, xbf, ROWS * EMB);
    for (int k = 0; k < 3; ++k) {
        k_cvt_bf16<<<1024, 256, 0, stream>>>(Wih[k], wihbf[k], GDIM * inK[k]);
        k_cvt_bf16<<<1024, 256, 0, stream>>>(Whh[k], whhbf[k], GDIM * HDIM);
        k_bias_sum<<<4, 256, 0, stream>>>(bih[k], bhh[k], bsum[k], GDIM);
    }
    k_init_barrier<<<1, 32, 0, stream>>>(bar);

    // ---- 3 layers: input GEMM + persistent scan ----
    const unsigned short* layerIn = xbf;
    unsigned short* layerOut[3] = {HoutA, HoutB, HoutA};
    for (int k = 0; k < 3; ++k) {
        // 512 M-groups x 16 N-groups = 8192 wave jobs, 8 waves/block -> 1024 blocks
        k_gemm_gin<<<1024, 256, 0, stream>>>(layerIn, wihbf[k], bsum[k], Gin, inK[k]);
        k_scan<<<NWG, 128, 0, stream>>>(Gin, whhbf[k], mask, hbuf, layerOut[k], bar);
        layerIn = layerOut[k];
    }

    // ---- MLP head ----
    k_mlp<<<ROWS / 256, 256, 0, stream>>>(layerOut[2], W1, b1, W2, b2, (float*)d_out);
}
